// GNN_75067438399962
// MI455X (gfx1250) — matrix operand order repaired and checked by reading, not compiled, not run on hardware
//
#include <hip/hip_runtime.h>
#include <hip/hip_bf16.h>
#include <stdint.h>

#define FDIM 128
#define NLAYERS 3
#define AGG_PITCH 1928   // 1920 cols + 8 pad halves -> 16B row skew for ds_load_b128
#define OUT_PITCH 136    // 128 cols + 8 pad halves
#define EDGE_PITCH 136   // 128 cols + 8 pad halves

typedef __attribute__((ext_vector_type(16))) __bf16 v16bf;
typedef __attribute__((ext_vector_type(8)))  float  v8f;

union ABFrag { uint4 u[2]; v16bf v; };

__device__ __forceinline__ unsigned short f2bf(float f) {
    unsigned u = __float_as_uint(f);
    u += 0x7FFFu + ((u >> 16) & 1u);              // round-to-nearest-even
    return (unsigned short)(u >> 16);
}
__device__ __forceinline__ float bf2f(unsigned short h) {
    return __uint_as_float(((unsigned)h) << 16);
}
// order-preserving float <-> uint mapping so integer atomicMin/Max implement float min/max
__device__ __forceinline__ unsigned ordEnc(float f) {
    unsigned u = __float_as_uint(f);
    return (u & 0x80000000u) ? ~u : (u | 0x80000000u);
}
__device__ __forceinline__ float ordDec(unsigned e) {
    unsigned u = (e & 0x80000000u) ? (e ^ 0x80000000u) : ~e;
    return __uint_as_float(u);
}
__device__ __forceinline__ v8f splat8(float b) { return (v8f){b, b, b, b, b, b, b, b}; }

__device__ __forceinline__ v8f wmma_bf16(ABFrag a, ABFrag b, v8f c) {
    return __builtin_amdgcn_wmma_f32_16x16x32_bf16(false, a.v, false, b.v, (short)0, c,
                                                   false, false);
}

// ---------------- small elementwise kernels ----------------

__global__ void k_zero_u32(unsigned* __restrict__ p, int n) {
    int i = blockIdx.x * blockDim.x + threadIdx.x;
    if (i < n) p[i] = 0u;
}

__global__ void k_count(const int* __restrict__ eidx, unsigned* __restrict__ cnt, int E) {
    int i = blockIdx.x * blockDim.x + threadIdx.x;
    if (i < E) atomicAdd(&cnt[eidx[E + i]], 1u);   // dst = edge_index[1]
}

__global__ void k_scalers(const unsigned* __restrict__ cnt, float* __restrict__ amp,
                          float* __restrict__ att, int N) {
    int i = blockIdx.x * blockDim.x + threadIdx.x;
    if (i >= N) return;
    float s = 0.f;
    #pragma unroll
    for (int k = 1; k <= 33; ++k) s += logf((float)k);   // PyG avg log-degree constant
    float avg = s * (1.f / 33.f);
    float deg = fmaxf((float)cnt[i], 1.f);
    float ld  = logf(deg + 1.f);
    amp[i] = ld / avg;
    att[i] = avg / ld;
}

__global__ void k_init_agg(float* __restrict__ s, float* __restrict__ q,
                           unsigned* __restrict__ mn, unsigned* __restrict__ mx, long n) {
    long i = (long)blockIdx.x * blockDim.x + threadIdx.x;
    if (i < n) { s[i] = 0.f; q[i] = 0.f; mn[i] = 0xFFFFFFFFu; mx[i] = 0u; }
}

__global__ void k_x2bf(const float* __restrict__ x, unsigned short* __restrict__ xa, long n) {
    long i = (long)blockIdx.x * blockDim.x + threadIdx.x;
    if (i < n) xa[i] = f2bf(x[i]);
}

// Pack a row-major fp32 (K x 128) matrix into bf16 WMMA B-fragment tiles.
// Tile t = kt*8+nt holds 512 halves; lane l stores its 16 operand halves contiguously.
__global__ void k_pack(const float* __restrict__ W, unsigned short* __restrict__ dst,
                       int nTiles) {
    int w    = (blockIdx.x * blockDim.x + threadIdx.x) >> 5;
    int lane = threadIdx.x & 31;
    if (w >= nTiles) return;
    int kt = w >> 3, nt = w & 7;
    int n  = lane & 15;
    int kb = kt * 32 + ((lane >= 16) ? 8 : 0);
    union { unsigned short s[16]; uint4 u[2]; } t;
    #pragma unroll
    for (int j = 0; j < 8; ++j) {
        t.s[2 * j]     = f2bf(W[(long)(kb + 2 * j + (j >= 4 ? 8 : 0))     * FDIM + nt * 16 + n]);
        t.s[2 * j + 1] = f2bf(W[(long)(kb + 2 * j + 1 + (j >= 4 ? 8 : 0)) * FDIM + nt * 16 + n]);
    }
    uint4* o = (uint4*)(dst + (long)w * 512 + lane * 16);
    o[0] = t.u[0];
    o[1] = t.u[1];
}

// ---------------- edge messages: h = cat([x_dst, x_src]) @ W_pre + b; fused scatter -----------
// Block = 8 waves; wave w owns output N-tile w so its 8 B tiles (64 VGPRs) stay register
// resident (no LICM spill). Block cooperatively stages the 16 dst + 16 src rows in LDS.

__global__ void __launch_bounds__(256) k_edge(
    const unsigned short* __restrict__ xa, const int* __restrict__ eidx,
    const unsigned short* __restrict__ wpre, const float* __restrict__ bpre,
    float* __restrict__ gsum, float* __restrict__ gsq,
    unsigned* __restrict__ gmn, unsigned* __restrict__ gmx, int E, int nTiles) {
    __shared__ unsigned short ax[32 * EDGE_PITCH];   // rows 0-15: x[dst], rows 16-31: x[src]
    __shared__ int ldsDst[16];

    int tid  = threadIdx.x;
    int lane = tid & 31;
    int wv   = tid >> 5;            // N-tile owned by this wave
    int r    = lane & 15;
    int klo  = (lane >= 16) ? 8 : 0;
    int hi8  = (lane >= 16) ? 8 : 0;

    // Register-resident B fragments: tiles kt*8 + wv, kt = 0..7 (dst half then src half)
    ABFrag B[8];
    #pragma unroll
    for (int kt = 0; kt < 8; ++kt) {
        const uint4* bp = (const uint4*)(wpre + (long)(kt * 8 + wv) * 512 + lane * 16);
        B[kt].u[0] = bp[0];
        B[kt].u[1] = bp[1];
    }
    float bias = bpre[wv * 16 + r];

    int srow   = tid >> 3;          // staging: row 0..31
    int schunk = (tid & 7) * 16;    // halves offset within row (32B chunks)

    for (int t = blockIdx.x; t < nTiles; t += gridDim.x) {
        int e0 = t * 16;
        // gather 32 feature rows into LDS (coalesced b128 global loads -> ds stores)
        {
            int e = e0 + (srow & 15);
            if (e >= E) e = E - 1;
            int node = eidx[(srow < 16 ? E : 0) + e];
            const uint4* s = (const uint4*)(xa + (long)node * FDIM + schunk);
            uint4* d       = (uint4*)(ax + srow * EDGE_PITCH + schunk);
            *d = *s;
            if (tid < 16) {
                int ee = e0 + tid;
                if (ee >= E) ee = E - 1;
                ldsDst[tid] = eidx[E + ee];
            }
        }
        int tn = t + gridDim.x;     // prefetch next tile's indices
        if (tn < nTiles) {
            __builtin_prefetch(eidx + tn * 16, 0, 0);
            __builtin_prefetch(eidx + E + tn * 16, 0, 0);
        }
        __syncthreads();

        v8f acc = splat8(bias);
        #pragma unroll
        for (int kt = 0; kt < 4; ++kt) {            // x_dst x W_pre[0:128]
            ABFrag a;
            const unsigned short* p = ax + r * EDGE_PITCH + kt * 32 + klo;
            a.u[0] = *(const uint4*)p;
            a.u[1] = *(const uint4*)(p + 16);
            acc = wmma_bf16(a, B[kt], acc);
        }
        #pragma unroll
        for (int kt = 0; kt < 4; ++kt) {            // x_src x W_pre[128:256]
            ABFrag a;
            const unsigned short* p = ax + (16 + r) * EDGE_PITCH + kt * 32 + klo;
            a.u[0] = *(const uint4*)p;
            a.u[1] = *(const uint4*)(p + 16);
            acc = wmma_bf16(a, B[4 + kt], acc);
        }
        // C layout: VGPR rr = row m = rr + (lane>=16 ? 8 : 0), col = wv*16 + (lane&15)
        #pragma unroll
        for (int rr = 0; rr < 8; ++rr) {
            int m    = rr + hi8;
            long off = (long)ldsDst[m] * FDIM + wv * 16 + r;
            float h  = acc[rr];
            atomicAdd(gsum + off, h);
            atomicAdd(gsq + off, h * h);
            unsigned oe = ordEnc(h);
            atomicMin(gmn + off, oe);
            atomicMax(gmx + off, oe);
        }
        __syncthreads();
    }
}

// ---------------- node update: post_nn + lin + relu, one 16-node tile per block ----------------

__global__ void __launch_bounds__(128) k_node(
    const unsigned short* __restrict__ xa, unsigned short* __restrict__ xn,
    const float* __restrict__ gsum, const float* __restrict__ gsq,
    const unsigned* __restrict__ gmn, const unsigned* __restrict__ gmx,
    const unsigned* __restrict__ cnt, const float* __restrict__ amp,
    const float* __restrict__ att, const unsigned short* __restrict__ wpost,
    const float* __restrict__ bpost, const unsigned short* __restrict__ wlin,
    const float* __restrict__ blin, int N) {
    __shared__ unsigned short lds[16 * AGG_PITCH];   // 61.7 KB: scaled agg, then reused for out
    int base = blockIdx.x * 16;
    int tid  = threadIdx.x;

    // Stage [mean,sum,std,min,max] x {1, amp, att} = 16 x 1920 bf16 in LDS
    for (int idx = tid; idx < 16 * 640; idx += blockDim.x) {
        int row  = idx / 640;
        int c    = idx - row * 640;
        int node = base + row; if (node >= N) node = N - 1;
        unsigned cn = cnt[node];
        float deg   = fmaxf((float)cn, 1.f);
        int sub = c >> 7, f = c & 127;
        long off = (long)node * FDIM + f;
        float s = gsum[off], val;
        if (sub == 0) val = s / deg;
        else if (sub == 1) val = s;
        else if (sub == 2) {
            float mean = s / deg;
            float v = gsq[off] / deg - mean * mean;
            val = sqrtf(fmaxf(v, 0.f) + 1e-5f);
        } else if (sub == 3) val = cn ? ordDec(gmn[off]) : 0.f;
        else                 val = cn ? ordDec(gmx[off]) : 0.f;
        float a = amp[node], t = att[node];
        unsigned short* lr = lds + row * AGG_PITCH;
        lr[c]        = f2bf(val);
        lr[640 + c]  = f2bf(val * a);
        lr[1280 + c] = f2bf(val * t);
    }
    __syncthreads();

    int lane = tid & 31;
    int wv   = tid >> 5;
    int r    = lane & 15;
    int klo  = (lane >= 16) ? 8 : 0;
    int hi8  = (lane >= 16) ? 8 : 0;
    int nt0  = wv * 2;                              // each wave owns 2 N-tiles of the output

    v8f acc0 = splat8(bpost[nt0 * 16 + r]);
    v8f acc1 = splat8(bpost[(nt0 + 1) * 16 + r]);
    int arow = base + r; if (arow >= N) arow = N - 1;
    const unsigned short* xrow = xa + (long)arow * FDIM;
    const unsigned short* lrow = lds + r * AGG_PITCH;

    for (int kt = 0; kt < 64; ++kt) {               // K = 2048 = cat([x, scaled_agg])
        ABFrag a;
        if (kt < 4) {
            a.u[0] = *(const uint4*)(xrow + kt * 32 + klo);
            a.u[1] = *(const uint4*)(xrow + kt * 32 + klo + 16);
        } else {
            const unsigned short* p = lrow + (kt - 4) * 32 + klo;
            a.u[0] = *(const uint4*)p;
            a.u[1] = *(const uint4*)(p + 16);
        }
        ABFrag b0, b1;
        const uint4* bp0 = (const uint4*)(wpost + (long)(kt * 8 + nt0) * 512 + lane * 16);
        const uint4* bp1 = (const uint4*)(wpost + (long)(kt * 8 + nt0 + 1) * 512 + lane * 16);
        b0.u[0] = bp0[0]; b0.u[1] = bp0[1];
        b1.u[0] = bp1[0]; b1.u[1] = bp1[1];
        acc0 = wmma_bf16(a, b0, acc0);
        acc1 = wmma_bf16(a, b1, acc1);
    }
    __syncthreads();

    // stage intermediate 16x128 as bf16 (reuse agg LDS), then apply W_lin
    #pragma unroll
    for (int rr = 0; rr < 8; ++rr) {
        int m = rr + hi8;
        lds[m * OUT_PITCH + nt0 * 16 + r]       = f2bf(acc0[rr]);
        lds[m * OUT_PITCH + (nt0 + 1) * 16 + r] = f2bf(acc1[rr]);
    }
    __syncthreads();

    v8f o0 = splat8(blin[nt0 * 16 + r]);
    v8f o1 = splat8(blin[(nt0 + 1) * 16 + r]);
    #pragma unroll
    for (int kt = 0; kt < 4; ++kt) {
        ABFrag a;
        const unsigned short* p = lds + r * OUT_PITCH + kt * 32 + klo;
        a.u[0] = *(const uint4*)p;
        a.u[1] = *(const uint4*)(p + 16);
        ABFrag b0, b1;
        const uint4* bp0 = (const uint4*)(wlin + (long)(kt * 8 + nt0) * 512 + lane * 16);
        const uint4* bp1 = (const uint4*)(wlin + (long)(kt * 8 + nt0 + 1) * 512 + lane * 16);
        b0.u[0] = bp0[0]; b0.u[1] = bp0[1];
        b1.u[0] = bp1[0]; b1.u[1] = bp1[1];
        o0 = wmma_bf16(a, b0, o0);
        o1 = wmma_bf16(a, b1, o1);
    }
    #pragma unroll
    for (int rr = 0; rr < 8; ++rr) {
        int m = rr + hi8;
        int node = base + m;
        if (node < N) {
            xn[(long)node * FDIM + nt0 * 16 + r]       = f2bf(fmaxf(o0[rr], 0.f));
            xn[(long)node * FDIM + (nt0 + 1) * 16 + r] = f2bf(fmaxf(o1[rr], 0.f));
        }
    }
}

// ---------------- final projection F -> 1 ----------------

__global__ void k_final(const unsigned short* __restrict__ x, const float* __restrict__ wout,
                        const float* __restrict__ bout, float* __restrict__ out, int N) {
    int lane = threadIdx.x & 31;
    int w    = (blockIdx.x * blockDim.x + threadIdx.x) >> 5;
    int nw   = (gridDim.x * blockDim.x) >> 5;
    for (int node = w; node < N; node += nw) {
        float a = 0.f;
        #pragma unroll
        for (int j = 0; j < 4; ++j) {
            int f = lane + 32 * j;
            a += bf2f(x[(long)node * FDIM + f]) * wout[f];
        }
        #pragma unroll
        for (int s = 16; s > 0; s >>= 1) a += __shfl_xor(a, s, 32);
        if (lane == 0) out[node] = a + bout[0];
    }
}

// ---------------- host launcher ----------------

extern "C" void kernel_launch(void* const* d_in, const int* in_sizes, int n_in, void* d_out,
                              int out_size, void* d_ws, size_t ws_size, hipStream_t stream) {
    const float* x      = (const float*)d_in[0];
    const int*   eidx   = (const int*)d_in[1];
    const float* W_pre  = (const float*)d_in[2];
    const float* b_pre  = (const float*)d_in[3];
    const float* W_post = (const float*)d_in[4];
    const float* b_post = (const float*)d_in[5];
    const float* W_lin  = (const float*)d_in[6];
    const float* b_lin  = (const float*)d_in[7];
    const float* W_out  = (const float*)d_in[8];
    const float* b_out  = (const float*)d_in[9];

    const int N = in_sizes[0] / FDIM;
    const int E = in_sizes[1] / 2;
    const long NF = (long)N * FDIM;

    char* p = (char*)d_ws;
    auto carve = [&](size_t bytes) -> char* {
        char* r = p;
        p += (bytes + 255) & ~(size_t)255;
        return r;
    };
    unsigned*       cnt  = (unsigned*)carve((size_t)N * 4);
    float*          amp  = (float*)carve((size_t)N * 4);
    float*          att  = (float*)carve((size_t)N * 4);
    float*          gsum = (float*)carve((size_t)NF * 4);
    float*          gsq  = (float*)carve((size_t)NF * 4);
    unsigned*       gmn  = (unsigned*)carve((size_t)NF * 4);
    unsigned*       gmx  = (unsigned*)carve((size_t)NF * 4);
    unsigned short* xb0  = (unsigned short*)carve((size_t)NF * 2);
    unsigned short* xb1  = (unsigned short*)carve((size_t)NF * 2);
    unsigned short* wpreP  = (unsigned short*)carve((size_t)NLAYERS * 64 * 512 * 2);
    unsigned short* wpostP = (unsigned short*)carve((size_t)NLAYERS * 512 * 512 * 2);
    unsigned short* wlinP  = (unsigned short*)carve((size_t)NLAYERS * 32 * 512 * 2);

    // degrees + scalers + bf16 activations
    k_zero_u32<<<(N + 255) / 256, 256, 0, stream>>>(cnt, N);
    k_count<<<(E + 255) / 256, 256, 0, stream>>>(eidx, cnt, E);
    k_scalers<<<(N + 255) / 256, 256, 0, stream>>>(cnt, amp, att, N);
    k_x2bf<<<(int)((NF + 255) / 256), 256, 0, stream>>>(x, xb0, NF);

    // repack weights into WMMA B-fragment tiles (bf16)
    for (int l = 0; l < NLAYERS; ++l) {
        k_pack<<<(64 + 3) / 4, 128, 0, stream>>>(W_pre + (long)l * 256 * FDIM,
                                                 wpreP + (long)l * 64 * 512, 64);
        k_pack<<<(512 + 3) / 4, 128, 0, stream>>>(W_post + (long)l * 2048 * FDIM,
                                                  wpostP + (long)l * 512 * 512, 512);
        k_pack<<<(32 + 3) / 4, 128, 0, stream>>>(W_lin + (long)l * FDIM * FDIM,
                                                 wlinP + (long)l * 32 * 512, 32);
    }

    unsigned short* xc = xb0;
    unsigned short* xn = xb1;
    const int eTiles = (E + 15) / 16;
    for (int l = 0; l < NLAYERS; ++l) {
        k_init_agg<<<(int)((NF + 255) / 256), 256, 0, stream>>>(gsum, gsq, gmn, gmx, NF);
        k_edge<<<1024, 256, 0, stream>>>(xc, eidx, wpreP + (long)l * 64 * 512,
                                         b_pre + (long)l * FDIM, gsum, gsq, gmn, gmx, E,
                                         eTiles);
        k_node<<<(N + 15) / 16, 128, 0, stream>>>(
            xc, xn, gsum, gsq, gmn, gmx, cnt, amp, att, wpostP + (long)l * 512 * 512,
            b_post + (long)l * FDIM, wlinP + (long)l * 32 * 512, b_lin + (long)l * FDIM, N);
        unsigned short* t = xc; xc = xn; xn = t;
    }
    k_final<<<512, 256, 0, stream>>>(xc, W_out, b_out, (float*)d_out, N);
}
